// Codebook_51393578664435
// MI455X (gfx1250) — matrix operand order repaired and checked by reading, not compiled, not run on hardware
//
#include <hip/hip_runtime.h>

typedef __attribute__((ext_vector_type(16))) __bf16          v16bf;
typedef __attribute__((ext_vector_type(8)))  float           v8f;

#define D_DIM    512
#define K_CODES  1024
#define HW       4096
#define N_TOK    65536
#define ZQ_ELEMS 33554432
// exp(-dist/TEMP) = exp2((dmin - dist) * (1/TEMP) * log2(e)),  TEMP = 0.5
#define INV_TEMP_LOG2E 2.8853900817779268f

#define ZB_STRIDE 520   // bf16 elements per row: 512 + 8 pad -> conflict-friendly, 16B-aligned rows

struct BF16x16 { uint4 lo; uint4 hi; };   // 32 bytes -> bit_cast to v16bf

__device__ __forceinline__ unsigned short f2bf(float f) {
    unsigned u = __float_as_uint(f);
    unsigned r = u + 0x7FFFu + ((u >> 16) & 1u);   // round-to-nearest-even
    return (unsigned short)(r >> 16);
}

// ---------------------------------------------------------------- kernel 0
// codebook f32 -> bf16, per-code squared norm, zero the loss scalar.
__global__ void __launch_bounds__(256) prep_kernel(
    const float* __restrict__ cb, unsigned short* __restrict__ cbb,
    float* __restrict__ cnorm, float* __restrict__ qloss) {
    const int c   = blockIdx.x;
    const int tid = threadIdx.x;
    const float* row = cb + (size_t)c * D_DIM;
    unsigned short* orow = cbb + (size_t)c * D_DIM;
    float s = 0.f;
#pragma unroll
    for (int i = 0; i < 2; ++i) {
        int k = tid + i * 256;
        float v = row[k];
        orow[k] = f2bf(v);
        s += v * v;
    }
    __shared__ float sred[256];
    sred[tid] = s;
    __syncthreads();
    for (int st = 128; st > 0; st >>= 1) {
        if (tid < st) sred[tid] += sred[tid + st];
        __syncthreads();
    }
    if (tid == 0) cnorm[c] = sred[0];
    if (c == 0 && tid == 0) *qloss = 0.f;
}

// ---------------------------------------------------------------- kernel 1
// One block = 64 tokens x all 1024 codes, 512 threads (16 waves).
// Waves pair up per 128-column stripe: wave pair shares B fragments (second
// wave hits WGP$), halving L2 codebook traffic vs M=32. z tile converted to
// bf16 once in LDS; A fragments are pure ds_load_b128.
__global__ void __launch_bounds__(512) dist_softmax_kernel(
    const float* __restrict__ z, const unsigned short* __restrict__ cbb,
    const float* __restrict__ cnormg, float* __restrict__ prob,
    int* __restrict__ minidx) {
    __shared__ unsigned short zB[64 * ZB_STRIDE];   // 66,560 B bf16 tile [m][d]
    __shared__ float cnormS[K_CODES];
    __shared__ float znormS[64];
    __shared__ float red[512];
    __shared__ unsigned long long rowkey[64];
    __shared__ float rowsum[64];

    const int tid    = threadIdx.x;
    const int lane   = tid & 31;
    const int wv     = tid >> 5;            // 16 waves
    const int lm     = lane & 15;
    const int lhalf  = lane >> 4;
    const int mgroup = wv >> 3;             // M-half (rows 0-31 / 32-63)
    const int nbase  = (wv & 7) << 7;       // 128-column stripe

    const int n0 = blockIdx.x << 6;         // 64 tokens = one full w row
    const int bi = n0 >> 12;
    const int hi = (n0 >> 6) & 63;
    const size_t zbase = (size_t)bi * (D_DIM * HW) + (size_t)hi * 64;

    // ---- load z tile, convert once to bf16 into LDS, f32 norm partials ----
    {
        const int m  = tid & 63;
        const int d0 = (tid >> 6) << 1;     // 0,2,...,14
        float s = 0.f;
        for (int dd = d0; dd < D_DIM; dd += 16) {
            float v0 = z[zbase + (size_t)dd * HW + m];
            float v1 = z[zbase + (size_t)(dd + 1) * HW + m];
            s += v0 * v0 + v1 * v1;
            ((unsigned*)zB)[(m * ZB_STRIDE + dd) >> 1] =
                (unsigned)f2bf(v0) | ((unsigned)f2bf(v1) << 16);
        }
        red[tid] = s;
    }
    for (int i = tid; i < K_CODES; i += 512) cnormS[i] = cnormg[i];
    if (tid < 64) { rowkey[tid] = ~0ull; rowsum[tid] = 0.f; }
    __syncthreads();
    if (tid < 64) {
        float s = 0.f;
#pragma unroll
        for (int j = 0; j < 8; ++j) s += red[(j << 6) + tid];
        znormS[tid] = s;
    }
    __syncthreads();

    // ---- GEMM: (64 x 512) * (512 x 1024)^T in k-steps of 32 ----
    const v8f zero8 = {0.f, 0.f, 0.f, 0.f, 0.f, 0.f, 0.f, 0.f};
    v8f acc[2][8];
#pragma unroll
    for (int mt = 0; mt < 2; ++mt)
#pragma unroll
        for (int t = 0; t < 8; ++t) acc[mt][t] = zero8;

#pragma unroll 2
    for (int ks = 0; ks < 16; ++ks) {
        const int k0 = ks << 5;
        const int ka = k0 + (lhalf << 4);   // A half-wave K split (contiguous)
        const int kb = k0 + (lhalf << 4);   // B half-wave K split

        // A fragments for this wave's two M-tiles: two ds_load_b128 each
        v16bf afrag[2];
#pragma unroll
        for (int mt = 0; mt < 2; ++mt) {
            const unsigned short* pr =
                zB + ((mgroup << 5) + (mt << 4) + lm) * ZB_STRIDE;
            BF16x16 u;
            u.lo = *(const uint4*)(pr + ka);
            u.hi = *(const uint4*)(pr + ka + 8);
            afrag[mt] = __builtin_bit_cast(v16bf, u);
        }

        // B fragments in groups of 4; each fragment feeds both M-tiles.
#pragma unroll
        for (int g = 0; g < 2; ++g) {
            v16bf bf[4];
#pragma unroll
            for (int j = 0; j < 4; ++j) {
                const int n = nbase + (((g << 2) + j) << 4) + lm;
                bf[j] = *(const v16bf*)(cbb + (size_t)n * D_DIM + kb);
            }
#pragma unroll
            for (int j = 0; j < 4; ++j) {
                const int t = (g << 2) + j;
                acc[0][t] = __builtin_amdgcn_wmma_f32_16x16x32_bf16(
                    false, afrag[0], false, bf[j], (short)0, acc[0][t], false, false);
                acc[1][t] = __builtin_amdgcn_wmma_f32_16x16x32_bf16(
                    false, afrag[1], false, bf[j], (short)0, acc[1][t], false, false);
            }
        }
    }

    // ---- dist = |z|^2 + |c|^2 - 2*dot ; per-row argmin via packed ds_min_u64 ----
    const int mrow0 = (mgroup << 5) + (lhalf << 3);   // + (mt<<4) + r

    float zn[2][8];
#pragma unroll
    for (int mt = 0; mt < 2; ++mt)
#pragma unroll
        for (int r = 0; r < 8; ++r) zn[mt][r] = znormS[mrow0 + (mt << 4) + r];

#pragma unroll
    for (int t = 0; t < 8; ++t) {
        const int n = nbase + (t << 4) + lm;
        const float cn = cnormS[n];
#pragma unroll
        for (int mt = 0; mt < 2; ++mt)
#pragma unroll
            for (int r = 0; r < 8; ++r)
                acc[mt][t][r] = zn[mt][r] + cn - 2.f * acc[mt][t][r];
    }
#pragma unroll
    for (int mt = 0; mt < 2; ++mt)
#pragma unroll
        for (int r = 0; r < 8; ++r) {
            float dm = acc[mt][0][r];
            int   ni = nbase + lm;
#pragma unroll
            for (int t = 1; t < 8; ++t) {
                const float dv = acc[mt][t][r];
                if (dv < dm) { dm = dv; ni = nbase + (t << 4) + lm; }
            }
            const unsigned long long key =
                ((unsigned long long)__float_as_uint(dm) << 32) | (unsigned)ni;
            atomicMin(&rowkey[mrow0 + (mt << 4) + r], key);
        }
    __syncthreads();

    float dmin[2][8];
#pragma unroll
    for (int mt = 0; mt < 2; ++mt)
#pragma unroll
        for (int r = 0; r < 8; ++r)
            dmin[mt][r] = __uint_as_float(
                (unsigned)(rowkey[mrow0 + (mt << 4) + r] >> 32));

    // ---- stable softmax: exp + row sums ----
#pragma unroll
    for (int mt = 0; mt < 2; ++mt)
#pragma unroll
        for (int r = 0; r < 8; ++r) {
            float s = 0.f;
#pragma unroll
            for (int t = 0; t < 8; ++t) {
                const float e = exp2f((dmin[mt][r] - acc[mt][t][r]) * INV_TEMP_LOG2E);
                acc[mt][t][r] = e;
                s += e;
            }
            atomicAdd(&rowsum[mrow0 + (mt << 4) + r], s);
        }
    __syncthreads();

    float inv[2][8];
#pragma unroll
    for (int mt = 0; mt < 2; ++mt)
#pragma unroll
        for (int r = 0; r < 8; ++r)
            inv[mt][r] = 1.f / rowsum[mrow0 + (mt << 4) + r];

#pragma unroll
    for (int mt = 0; mt < 2; ++mt)
#pragma unroll
        for (int t = 0; t < 8; ++t)
#pragma unroll
            for (int r = 0; r < 8; ++r) {
                const int m = mrow0 + (mt << 4) + r;
                prob[(size_t)(n0 + m) * K_CODES + nbase + (t << 4) + lm] =
                    acc[mt][t][r] * inv[mt][r];
            }

    if (tid < 64) minidx[n0 + tid] = (int)(rowkey[tid] & 0xFFFFFFFFull);
}

// ---------------------------------------------------------------- kernel 2
// z_q gather in (b,d,h,w) layout, vectorized x4 along w (float4/int4), fused
// loss reduction with one global f32 atomic per block.
__global__ void __launch_bounds__(256) gather_loss_kernel(
    const float* __restrict__ z, const float* __restrict__ cb,
    const int* __restrict__ minidx, float* __restrict__ zq,
    float* __restrict__ qloss) {
    const int tid = threadIdx.x;
    const unsigned idx = (blockIdx.x * 256u + tid) * 4u;   // 4 consecutive w
    const int wi = idx & 63;
    const int hi = (idx >> 6) & 63;
    const int dd = (idx >> 12) & 511;
    const int bi = idx >> 21;
    const int token = (bi << 12) | (hi << 6) | wi;

    const int4 mi = *(const int4*)(minidx + token);
    float4 v;
    v.x = cb[(size_t)mi.x * D_DIM + dd];
    v.y = cb[(size_t)mi.y * D_DIM + dd];
    v.z = cb[(size_t)mi.z * D_DIM + dd];
    v.w = cb[(size_t)mi.w * D_DIM + dd];
    *(float4*)(zq + idx) = v;

    const float4 zv = *(const float4*)(z + idx);
    const float dx = v.x - zv.x, dy = v.y - zv.y, dz = v.z - zv.z, dw = v.w - zv.w;

    __shared__ float sred[256];
    sred[tid] = dx * dx + dy * dy + dz * dz + dw * dw;
    __syncthreads();
    for (int st = 128; st > 0; st >>= 1) {
        if (tid < st) sred[tid] += sred[tid + st];
        __syncthreads();
    }
    // q_loss = (1 + BETA) * sum(diff^2) / (n*d)
    if (tid == 0) atomicAdd(qloss, sred[0] * (1.25f / 33554432.f));
}

// ---------------------------------------------------------------- launcher
extern "C" void kernel_launch(void* const* d_in, const int* in_sizes, int n_in,
                              void* d_out, int out_size, void* d_ws, size_t ws_size,
                              hipStream_t stream) {
    const float* z  = (const float*)d_in[0];   // (16, 512, 64, 64) f32
    const float* cb = (const float*)d_in[1];   // (1024, 512)       f32

    // workspace: bf16 codebook (1 MB) | cnorm (4 KB) | min_idx (256 KB)
    unsigned short* cbb = (unsigned short*)d_ws;
    float* cnorm  = (float*)((char*)d_ws + (size_t)K_CODES * D_DIM * 2);
    int*   minidx = (int*)((char*)d_ws + (size_t)K_CODES * D_DIM * 2 + K_CODES * 4);

    float* out   = (float*)d_out;
    float* zq    = out;                        // 33,554,432
    float* qloss = out + ZQ_ELEMS;             // 1
    float* prob  = out + ZQ_ELEMS + 1;         // 67,108,864

    prep_kernel<<<K_CODES, 256, 0, stream>>>(cb, cbb, cnorm, qloss);
    dist_softmax_kernel<<<N_TOK / 64, 512, 0, stream>>>(z, cbb, cnorm, prob, minidx);
    gather_loss_kernel<<<ZQ_ELEMS / 1024, 256, 0, stream>>>(z, cb, minidx, zq, qloss);
}
